// BidirectionalLaCTSwiGLU_23012434772134
// MI455X (gfx1250) — matrix operand order repaired and checked by reading, not compiled, not run on hardware
//
#include <hip/hip_runtime.h>
#include <hip/hip_bf16.h>

// ---------------------------------------------------------------------------
// CDNA5 (gfx1250) bf16 WMMA building blocks. wave32, 16x16x32, f32 accumulate.
// Operand tiles live in fragment-native LDS order (lane half h holds k 16h..16h+15): fragment load
// = one 32B vector load (2x ds_load_b128). Global->LDS staging uses the async
// copy engine (GLOBAL_LOAD_ASYNC_TO_LDS_B128, ASYNCcnt) with per-lane swizzled
// LDS destinations, so tiles land directly in operand order.
// ---------------------------------------------------------------------------
typedef __attribute__((ext_vector_type(16))) __bf16 v16bf;
typedef __attribute__((ext_vector_type(8)))  float  v8f;

#define DEV static __device__ __forceinline__

DEV unsigned short f2bf(float f) {
    union { float f; unsigned u; } a; a.f = f;
    unsigned r = a.u + 0x7FFFu + ((a.u >> 16) & 1u);   // round-to-nearest-even
    return (unsigned short)(r >> 16);
}
DEV float bf2f(unsigned short s) {
    union { unsigned u; float f; } a; a.u = ((unsigned)s) << 16;
    return a.f;
}
DEV float sigmoidf_(float x) { return 1.0f / (1.0f + __expf(-x)); }

DEV v8f zero8() { v8f z = {0.f,0.f,0.f,0.f,0.f,0.f,0.f,0.f}; return z; }

DEV v8f wmma_bf(v16bf a, v16bf b, v8f c) {
    return __builtin_amdgcn_wmma_f32_16x16x32_bf16(false, a, false, b,
                                                   (short)0, c, false, false);
}

// ---- CDNA5 async copy: global -> LDS, per-lane LDS byte address ------------
DEV void async_b128(unsigned lds_byte, const void* g) {
    asm volatile("global_load_async_to_lds_b128 %0, %1, off"
                 :: "v"(lds_byte), "v"((unsigned long long)g)
                 : "memory");
}
DEV void wait_async0() { asm volatile("s_wait_asynccnt 0" ::: "memory"); }

// ---- fragment-native element index helpers ---------------------------------
// A operand (16x32 bf16): lane = (m&15) + 16*((k>>4)&1); slot = k&15
DEV int a_idx(int mt_total, int m, int k) {
    return (((k >> 5) * mt_total + (m >> 4)) * 32 + (m & 15) + (((k >> 4) & 1) << 4)) * 16
           + (k & 15);
}
// B operand (32x16 bf16): lane = (n&15) + 16*((k>>4)&1); slot = k&15
DEV int b_idx(int nt_total, int k, int n) {
    return (((k >> 5) * nt_total + (n >> 4)) * 32 + (n & 15) + (((k >> 4) & 1) << 4)) * 16
           + (k & 15);
}
// C/D (16x16): lane = (n&15) + 16*((m>>3)&1); slot = m&7
DEV int c_idx(int nt_total, int m, int n) {
    return (((m >> 4) * nt_total + (n >> 4)) * 32 + (n & 15) + (((m >> 3) & 1) << 4)) * 8
           + (m & 7);
}

// ---- vector fragment load/store on fragment-native LDS --------------------
DEV v16bf frag_ld16(const unsigned short* P, int tile) {
    return *(const v16bf*)(P + (tile * 32 + (threadIdx.x & 31)) * 16);
}
DEV v8f frag_ldc(const float* P, int tile) {
    return *(const v8f*)(P + (tile * 32 + (threadIdx.x & 31)) * 8);
}
DEV void frag_stc(float* P, int tile, v8f c) {
    *(v8f*)(P + (tile * 32 + (threadIdx.x & 31)) * 8) = c;
}
DEV void frag_stc_bf(unsigned short* P, int tile, v8f c) {   // C-native bf16
    unsigned short* p = P + (tile * 32 + (threadIdx.x & 31)) * 8;
#pragma unroll
    for (int j = 0; j < 4; ++j) {
        unsigned u = (unsigned)f2bf(c[2*j]) | ((unsigned)f2bf(c[2*j+1]) << 16);
        *(unsigned*)(p + 2*j) = u;
    }
}

// ---- gather loaders with paired (uint) reads: contiguous-pair cases only --
DEV v16bf frag_a_u(const unsigned short* A, int lda, int row0, int k0) {
    int l = threadIdx.x & 31; int m = row0 + (l & 15); int h8 = (l >> 4) << 3;
    union { v16bf v; unsigned u[8]; } u;
#pragma unroll
    for (int i = 0; i < 8; ++i) {
        int k = k0 + 2 * i + (h8 << 1);
        u.u[i] = *(const unsigned*)&A[m * lda + k];
    }
    return u.v;
}
DEV v16bf frag_bt_u(const unsigned short* S, int lds_, int k0, int col0) {
    int l = threadIdx.x & 31; int n = col0 + (l & 15); int h16 = (l >> 4) << 4;
    union { v16bf v; unsigned u[8]; } u;
#pragma unroll
    for (int i = 0; i < 8; ++i) {
        int k = k0 + 2 * i + h16;
        u.u[i] = *(const unsigned*)&S[n * lds_ + k];
    }
    return u.v;
}
DEV v16bf frag_a_g(const float* A, int lda, int row0, int k0) {
    int l = threadIdx.x & 31; int m = row0 + (l & 15); int h8 = (l >> 4) << 3;
    union { v16bf v; unsigned short s[16]; } u;
#pragma unroll
    for (int i = 0; i < 8; ++i) {
        int k = k0 + 2 * i + (h8 << 1);
        u.s[2*i]   = f2bf(A[m*lda + k]);
        u.s[2*i+1] = f2bf(A[m*lda + k + 1]);
    }
    return u.v;
}
DEV v16bf frag_a_gt(const float* S, int lds_, int row0, int k0) {
    int l = threadIdx.x & 31; int m = row0 + (l & 15); int h8 = (l >> 4) << 3;
    union { v16bf v; unsigned short s[16]; } u;
#pragma unroll
    for (int i = 0; i < 8; ++i) {
        int k = k0 + 2 * i + (h8 << 1);
        u.s[2*i]   = f2bf(S[k*lds_ + m]);
        u.s[2*i+1] = f2bf(S[(k+1)*lds_ + m]);
    }
    return u.v;
}

// ---------------------------------------------------------------------------
// Problem constants
// ---------------------------------------------------------------------------
#define DIM_    2048
#define L_      4096
#define BH_     32
#define NH_     16
#define HD_     128
#define ROWS_   8192          // B*L
#define QKVC_   6144          // 3*DIM

// ---------------------------------------------------------------------------
// Kernel 0: f32 -> bf16 cast
// ---------------------------------------------------------------------------
__global__ void k_cast(const float* __restrict__ src, unsigned short* __restrict__ dst, int n) {
    int i = blockIdx.x * blockDim.x + threadIdx.x;
    if (i < n) dst[i] = f2bf(src[i]);
}

// ---------------------------------------------------------------------------
// Kernel 1: lr = softplus(x @ w_lr^T + BASE_LR_INV) -> [3][32][4096]
// ---------------------------------------------------------------------------
__global__ void __launch_bounds__(256) k_lr(const float* __restrict__ x,
                                            const float* __restrict__ w_lr,
                                            float* __restrict__ lr) {
    int wave = blockIdx.x * 8 + (threadIdx.x >> 5);
    int lane = threadIdx.x & 31;
    int r = wave / 48, j = wave % 48;
    if (r >= ROWS_) return;
    const float* xr = x + (size_t)r * DIM_;
    const float* wr = w_lr + (size_t)j * DIM_;
    float s = 0.f;
    for (int t = lane; t < DIM_; t += 32) s += xr[t] * wr[t];
#pragma unroll
    for (int off = 16; off; off >>= 1) s += __shfl_down(s, off);
    if (lane == 0) {
        float z = s + (-4.6001660142815113f);   // BASE_LR_INV
        float sp = (z > 20.f) ? z : log1pf(__expf(z));
        int sidx = j % 3, h = j / 3, b = r >> 12, l = r & 4095;
        lr[((size_t)(sidx * BH_ + b * NH_ + h)) * L_ + l] = sp;
    }
}

// ---------------------------------------------------------------------------
// Kernel 2: qkv = silu(x @ w_qkv^T); q,k row-l2-normalized; bf16 [3][32][L][128]
// A tile async-staged A-native from x; B tile async-staged B-native straight
// from ROW-MAJOR w_qkv (row n has contiguous k -> native slot runs).
// Dynamic LDS: Als @ byte 0, Bls @ byte 8192.
// ---------------------------------------------------------------------------
__global__ void __launch_bounds__(256) k_qkv(const unsigned short* __restrict__ xb,
                                             const unsigned short* __restrict__ wb,
                                             unsigned short* __restrict__ qkvh) {
    extern __shared__ char sm[];
    unsigned short* Als = (unsigned short*)sm;            // A-native, 8 m-tiles
    unsigned short* Bls = (unsigned short*)(sm + 8192);   // B-native, 8 n-tiles
    int m0 = blockIdx.x * 128;
    int nb = blockIdx.y;                 // 0..47 -> (s, h)
    int n0 = nb * 128;
    int tid = threadIdx.x, w = tid >> 5;
    v8f acc[8];
#pragma unroll
    for (int t = 0; t < 8; ++t) acc[t] = zero8();

    for (int kt = 0; kt < DIM_; kt += 32) {
#pragma unroll
        for (int p = 0; p < 2; ++p) {                     // 512 b128s per tile, 1024 per tile pair
            int i = tid + p * 256;
            int m = i >> 2, oct = i & 3, k8 = oct * 8;
            unsigned adst = 2u * ((unsigned)(((m >> 4) * 32 + (m & 15) + ((oct >> 1) << 4)) * 16
                                             + ((oct & 1) << 3)));
            async_b128(adst, &xb[(size_t)(m0 + m) * DIM_ + kt + k8]);
            unsigned bdst = 8192u + 2u * ((unsigned)(((m >> 4) * 32 + (m & 15) + ((oct >> 1) << 4)) * 16
                                                     + ((oct & 1) << 3)));
            async_b128(bdst, &wb[(size_t)(n0 + m) * DIM_ + kt + k8]);
        }
        if (kt + 32 < DIM_) {
            __builtin_prefetch(&xb[(size_t)(m0 + (tid >> 1)) * DIM_ + kt + 32]);
            __builtin_prefetch(&wb[(size_t)(n0 + (tid >> 1)) * DIM_ + kt + 32]);
        }
        wait_async0();
        __syncthreads();
        v16bf a = frag_ld16(Als, w);
#pragma unroll
        for (int tc = 0; tc < 8; ++tc)
            acc[tc] = wmma_bf(a, frag_ld16(Bls, tc), acc[tc]);
        __syncthreads();
    }

    // fused silu + row l2-norm, scatter to per-head bf16 layout
    int s_ = nb >> 4, h = nb & 15;
    int lI = tid & 31, n_ = lI & 15, r8 = (lI >> 4) << 3;
    float rowss[8];
#pragma unroll
    for (int j = 0; j < 8; ++j) rowss[j] = 0.f;
#pragma unroll
    for (int tc = 0; tc < 8; ++tc)
#pragma unroll
        for (int j = 0; j < 8; ++j) {
            float v = acc[tc][j];
            v = v * sigmoidf_(v);
            acc[tc][j] = v;
            rowss[j] += v * v;
        }
#pragma unroll
    for (int j = 0; j < 8; ++j) {
#pragma unroll
        for (int m = 8; m; m >>= 1) rowss[j] += __shfl_xor(rowss[j], m);
    }
#pragma unroll
    for (int j = 0; j < 8; ++j) {
        int brow = m0 + w * 16 + r8 + j;
        int b = brow >> 12, l = brow & 4095;
        float inv = (s_ < 2) ? 1.f / (sqrtf(rowss[j]) + 1e-5f) : 1.f;
        size_t base = ((size_t)(s_ * BH_ + b * NH_ + h) * L_ + l) * HD_;
#pragma unroll
        for (int tc = 0; tc < 8; ++tc)
            qkvh[base + tc * 16 + n_] = f2bf(acc[tc][j] * inv);
    }
}

// ---------------------------------------------------------------------------
// Kernel 3: per-head LaCT. 1 block/head, 256 thr, ~292KB dynamic LDS.
// ---------------------------------------------------------------------------
#define DW_BYTES   (3 * 16384 * 4)       // 196608: C-native f32 accumulators
#define SCR_BYTES  (3 * 16384 * 2)       // 98304:  NS buffers / phase staging
#define MISC_FLTS  (96 + 384 + 256 + 160)
#define SMEM_TOTAL (DW_BYTES + SCR_BYTES + MISC_FLTS * 4)

__global__ void __launch_bounds__(256) k_head(const unsigned short* __restrict__ qkvh,
                                              const float* __restrict__ lr,
                                              const float* __restrict__ w0g,
                                              const float* __restrict__ w1g,
                                              const float* __restrict__ w2g,
                                              const float* __restrict__ onw,
                                              unsigned short* __restrict__ onb) {
    extern __shared__ char smem[];
    float* DW  = (float*)smem;                            // 3 x 16384 f32 (C-native)
    char*  scr = smem + DW_BYTES;
    float* lr3 = (float*)(scr + SCR_BYTES);               // 96
    float* wn  = lr3 + 96;                                // 384
    float* red = wn + 384;                                // 256
    float* aux = red + 256;                               // 160

    int hb = blockIdx.x, h = hb & 15;
    int tid = threadIdx.x, w = tid >> 5;
    const float* W0 = w0g + (size_t)h * 16384;
    const float* W1 = w1g + (size_t)h * 16384;
    const float* W2 = w2g + (size_t)h * 16384;

    for (int i = tid; i < 3 * 16384; i += 256) DW[i] = 0.f;
    for (int row = tid; row < 384; row += 256) {
        int mi = row >> 7, r = row & 127;
        const float* Wp = (mi == 0 ? W0 : (mi == 1 ? W1 : W2)) + r * 128;
        float ss = 0.f;
        for (int c2 = 0; c2 < 128; ++c2) ss += Wp[c2] * Wp[c2];
        wn[row] = sqrtf(ss);
    }
    __syncthreads();

    // ---------------- phase 1: accumulate dw0/dw1/dw2 ----------------------
    unsigned short* scr16 = (unsigned short*)scr;         // 4096-ushort slabs
    unsigned short* KB2 = scr16 + 2 * 4096;     // k as B[k=l][n=d]  (1kt x 8nt)
    unsigned short* VA  = scr16 + 3 * 4096;     // v as A[m=d][k=l]  (8mt x 1kt)
    unsigned short* GT  = scr16 + 4 * 4096;     // gate raw  (C-native, 2nt)
    unsigned short* HR  = scr16 + 5 * 4096;     // hidden raw
    unsigned short* DH  = scr16 + 6 * 4096;     // dhidden
    unsigned short* HS  = scr16 + 7 * 4096;     // hidden*lr1 as B[k=l][n=m]
    unsigned short* DGA = scr16 + 8 * 4096;     // dgate_pre*lr0 as A[m][k=l]
    unsigned short* DHB = scr16 + 9 * 4096;     // dhid_bm*lr2 as A[m][k=l]
    unsigned short* KB1 = scr16;                // k as B[k=d][n=l]  (4kt x 2nt)
    unsigned short* VB1 = scr16 + 1 * 4096;     // v as B[k=d][n=l]
    const unsigned KB1_OFF = (unsigned)DW_BYTES;          // LDS byte offsets
    const unsigned VB1_OFF = (unsigned)DW_BYTES + 8192u;

    v16bf a0[4], a1t[4], a2[4];                 // persistent band frags of w0, w1^T, w2
#pragma unroll
    for (int ks = 0; ks < 4; ++ks) {
        a0[ks]  = frag_a_g (W0, 128, w * 16, ks * 32);
        a2[ks]  = frag_a_g (W2, 128, w * 16, ks * 32);
        a1t[ks] = frag_a_gt(W1, 128, w * 16, ks * 32);
    }

    const unsigned short* Qg = qkvh + (size_t)(0 * BH_ + hb) * L_ * HD_;
    const unsigned short* Kg = qkvh + (size_t)(1 * BH_ + hb) * L_ * HD_;
    const unsigned short* Vg = qkvh + (size_t)(2 * BH_ + hb) * L_ * HD_;
    float* dw0 = DW;
    float* dw1 = DW + 16384;
    float* dw2 = DW + 32768;

    for (int c = 0; c < 128; ++c) {                       // Lc = 32 chunks
        // async-stage KB1/VB1 directly in B-native order (slot runs along d)
#pragma unroll
        for (int p = 0; p < 2; ++p) {
            int i = tid + p * 256;
            int l = i >> 4, oct = i & 15, d8 = oct * 8;
            unsigned dst = 2u * ((unsigned)((((d8 >> 5) * 2 + (l >> 4)) * 32 + (l & 15)
                                            + (((d8 >> 4) & 1) << 4)) * 16 + (d8 & 15)));
            async_b128(KB1_OFF + dst, &Kg[(size_t)(c * 32 + l) * HD_ + d8]);
            async_b128(VB1_OFF + dst, &Vg[(size_t)(c * 32 + l) * HD_ + d8]);
        }
        // VGPR path for the element-scattered layouts (KB2, VA)
        for (int i = tid; i < 2048; i += 256) {
            int l = i >> 6, d = (i & 63) << 1;
            unsigned ku = *(const unsigned*)&Kg[(size_t)(c * 32 + l) * HD_ + d];
            unsigned vu = *(const unsigned*)&Vg[(size_t)(c * 32 + l) * HD_ + d];
            KB2[b_idx(8, l, d)]     = (unsigned short)ku;
            KB2[b_idx(8, l, d + 1)] = (unsigned short)(ku >> 16);
            VA [a_idx(8, d, l)]     = (unsigned short)vu;
            VA [a_idx(8, d + 1, l)] = (unsigned short)(vu >> 16);
        }
        if (tid < 96) {
            int s_ = tid >> 5, i2 = tid & 31;
            lr3[tid] = lr[((size_t)(s_ * BH_ + hb)) * L_ + c * 32 + i2];
        }
        wait_async0();
        __syncthreads();

        // gate = w0 k^T ; hraw = w2 k^T ; dh = w1^T v^T   -> [128][32]
#pragma unroll
        for (int tl = 0; tl < 2; ++tl) {
            v8f g = zero8(), hh = zero8(), dd = zero8();
#pragma unroll
            for (int ks = 0; ks < 4; ++ks) {
                v16bf bk = frag_ld16(KB1, ks * 2 + tl);
                v16bf bv = frag_ld16(VB1, ks * 2 + tl);
                g  = wmma_bf(a0[ks],  bk, g);
                hh = wmma_bf(a2[ks],  bk, hh);
                dd = wmma_bf(a1t[ks], bv, dd);
            }
            frag_stc_bf(GT, w * 2 + tl, g);
            frag_stc_bf(HR, w * 2 + tl, hh);
            frag_stc_bf(DH, w * 2 + tl, dd);
        }
        __syncthreads();

        // elementwise SwiGLU backprop; fold lr into A-side operands
        for (int i = tid; i < 4096; i += 256) {
            int m = i >> 5, l = i & 31;
            float gv = bf2f(GT[c_idx(2, m, l)]);
            float hr = bf2f(HR[c_idx(2, m, l)]);
            float dv = bf2f(DH[c_idx(2, m, l)]);
            float sg = sigmoidf_(gv);
            float silu_g = gv * sg;
            float dga = dv * hr * sg * (1.f + gv * (1.f - sg));
            HS [b_idx(8, l, m)] = f2bf(silu_g * hr * lr3[32 + l]);
            DHB[a_idx(8, m, l)] = f2bf(dv * silu_g * lr3[64 + l]);
            DGA[a_idx(8, m, l)] = f2bf(dga * lr3[l]);
        }
        __syncthreads();

        // dw0 += DGA@K ; dw2 += DHB@K ; dw1 += V^T @ HS^T  (C in/out LDS)
        {
            v16bf aA = frag_ld16(DGA, w);
            v16bf aB = frag_ld16(DHB, w);
            v16bf aV = frag_ld16(VA, w);
#pragma unroll
            for (int tc = 0; tc < 8; ++tc) {
                v16bf bk = frag_ld16(KB2, tc);
                frag_stc(dw0, w * 8 + tc, wmma_bf(aA, bk, frag_ldc(dw0, w * 8 + tc)));
                frag_stc(dw2, w * 8 + tc, wmma_bf(aB, bk, frag_ldc(dw2, w * 8 + tc)));
                v16bf bh = frag_ld16(HS, tc);
                frag_stc(dw1, w * 8 + tc, wmma_bf(aV, bh, frag_ldc(dw1, w * 8 + tc)));
            }
        }
        __syncthreads();
    }

    // ---------------- phase 2: Newton-Schulz (5 it, bf16) -------------------
    unsigned short* X  = scr16;                 // row-major [128][128]
    unsigned short* A_ = scr16 + 16384;         // A / X^T scratch
    unsigned short* Bm = scr16 + 32768;

    for (int mi = 0; mi < 3; ++mi) {
        float* G = DW + mi * 16384;
        float ss = 0.f;                                   // Frobenius norm (perm-invariant)
        for (int i = tid; i < 16384; i += 256) { float v = G[i]; ss += v * v; }
        red[tid] = ss; __syncthreads();
        for (int s2 = 128; s2 > 0; s2 >>= 1) {
            if (tid < s2) red[tid] += red[tid + s2];
            __syncthreads();
        }
        float inv = 1.f / (sqrtf(red[0]) + 1e-7f);
        for (int i = tid; i < 16384; i += 256) {
            int r = i >> 7, cc = i & 127;
            X[r * 128 + cc] = f2bf(G[c_idx(8, r, cc)] * inv);
        }
        __syncthreads();

        const float NSC[5][3] = {{4.0848f, -6.8946f, 2.927f},
                                 {3.9505f, -6.3029f, 2.6377f},
                                 {3.7418f, -5.5913f, 2.3037f},
                                 {2.8769f, -3.1427f, 1.2046f},
                                 {2.8366f, -3.0525f, 1.2012f}};
#pragma unroll
        for (int it = 0; it < 5; ++it) {
            float ca = NSC[it][0], cb = NSC[it][1], cc_ = NSC[it][2];
            int lI = tid & 31, nn = lI & 15, r8 = (lI >> 4) << 3;
            // A = X X^T  (B[k][n] = X[n][k]: contiguous pairs)
#pragma unroll
            for (int tc = 0; tc < 8; ++tc) {
                v8f acc = zero8();
#pragma unroll
                for (int ks = 0; ks < 4; ++ks)
                    acc = wmma_bf(frag_a_u(X, 128, w * 16, ks * 32),
                                  frag_bt_u(X, 128, ks * 32, tc * 16), acc);
#pragma unroll
                for (int j = 0; j < 8; ++j)
                    A_[(w * 16 + r8 + j) * 128 + tc * 16 + nn] = f2bf(acc[j]);
            }
            __syncthreads();
            // Bm = cb*A + cc*(A@A)   (A symmetric -> B-op via frag_bt)
#pragma unroll
            for (int tc = 0; tc < 8; ++tc) {
                v8f acc = zero8();
#pragma unroll
                for (int ks = 0; ks < 4; ++ks)
                    acc = wmma_bf(frag_a_u(A_, 128, w * 16, ks * 32),
                                  frag_bt_u(A_, 128, ks * 32, tc * 16), acc);
#pragma unroll
                for (int j = 0; j < 8; ++j) {
                    int r = w * 16 + r8 + j, n2 = tc * 16 + nn;
                    Bm[r * 128 + n2] = f2bf(cb * bf2f(A_[r * 128 + n2]) + cc_ * acc[j]);
                }
            }
            __syncthreads();
            // X^T into A_ (A dead)
            for (int i = tid; i < 8192; i += 256) {
                int r = i >> 6, c2 = (i & 63) << 1;
                unsigned u = *(const unsigned*)&X[r * 128 + c2];
                A_[c2 * 128 + r]       = (unsigned short)u;
                A_[(c2 + 1) * 128 + r] = (unsigned short)(u >> 16);
            }
            __syncthreads();
            // X = ca*X + Bm@X  (B-op via X^T; in-place, lane-owned elements)
#pragma unroll
            for (int tc = 0; tc < 8; ++tc) {
                v8f acc = zero8();
#pragma unroll
                for (int ks = 0; ks < 4; ++ks)
                    acc = wmma_bf(frag_a_u(Bm, 128, w * 16, ks * 32),
                                  frag_bt_u(A_, 128, ks * 32, tc * 16), acc);
#pragma unroll
                for (int j = 0; j < 8; ++j) {
                    int r = w * 16 + r8 + j, n2 = tc * 16 + nn;
                    X[r * 128 + n2] = f2bf(ca * bf2f(X[r * 128 + n2]) + acc[j]);
                }
            }
            __syncthreads();
        }

        // w_final = row_normalize(X + G) * wn ; stored A-native over G region
        if (tid < 128) {
            float ss2 = 0.f;
            for (int c2 = 0; c2 < 128; ++c2) {
                float v = bf2f(X[tid * 128 + c2]) + G[c_idx(8, tid, c2)];
                ss2 += v * v;
            }
            aux[tid] = wn[mi * 128 + tid] / (sqrtf(ss2) + 1e-5f);
        }
        __syncthreads();
        unsigned short outv[64];
        int cnt = 0;
        for (int i = tid; i < 16384; i += 256) {
            int r = i >> 7, c2 = i & 127;
            outv[cnt++] = f2bf((bf2f(X[r * 128 + c2]) + G[c_idx(8, r, c2)]) * aux[r]);
        }
        __syncthreads();
        unsigned short* WF = (unsigned short*)G;          // A-native final weight
        cnt = 0;
        for (int i = tid; i < 16384; i += 256) {
            int r = i >> 7, c2 = i & 127;
            WF[a_idx(8, r, c2)] = outv[cnt++];
        }
        __syncthreads();
    }

    // ---------------- phase 3: forward with updated weights -----------------
    const unsigned short* F0 = (const unsigned short*)(DW);
    const unsigned short* F1 = (const unsigned short*)(DW + 16384);
    const unsigned short* F2 = (const unsigned short*)(DW + 32768);
    v16bf f0[4], f1[4], f2_[4];
#pragma unroll
    for (int ks = 0; ks < 4; ++ks) {
        f0[ks]  = frag_ld16(F0, ks * 8 + w);
        f1[ks]  = frag_ld16(F1, ks * 8 + w);
        f2_[ks] = frag_ld16(F2, ks * 8 + w);
    }
    unsigned short* QB1 = scr16;                // q as B[k=d][n=l] (4kt x 2nt)
    unsigned short* GH  = scr16 + 4096;         // silu(g)*h as B[k=m][n=l]
    float* OF = (float*)(scr16 + 8192);         // o chunk C-native (8mt x 2nt)
    float* colsc = aux + 128;
    const unsigned QB1_OFF = (unsigned)DW_BYTES;
    int b_ = hb >> 4, hh2 = hb & 15;

    for (int c = 0; c < 128; ++c) {
#pragma unroll
        for (int p = 0; p < 2; ++p) {
            int i = tid + p * 256;
            int l = i >> 4, oct = i & 15, d8 = oct * 8;
            unsigned dst = 2u * ((unsigned)((((d8 >> 5) * 2 + (l >> 4)) * 32 + (l & 15)
                                            + (((d8 >> 4) & 1) << 4)) * 16 + (d8 & 15)));
            async_b128(QB1_OFF + dst, &Qg[(size_t)(c * 32 + l) * HD_ + d8]);
        }
        wait_async0();
        __syncthreads();
        int lI = tid & 31, nn = lI & 15, r8 = (lI >> 4) << 3;
#pragma unroll
        for (int tl = 0; tl < 2; ++tl) {
            v8f g = zero8(), hh = zero8();
#pragma unroll
            for (int ks = 0; ks < 4; ++ks) {
                v16bf bq = frag_ld16(QB1, ks * 2 + tl);
                g  = wmma_bf(f0[ks],  bq, g);
                hh = wmma_bf(f2_[ks], bq, hh);
            }
            int base = b_idx(2, w * 16 + r8, tl * 16 + nn);
#pragma unroll
            for (int j = 0; j < 4; ++j) {
                float g0 = g[2*j],   h0 = hh[2*j];
                float g1 = g[2*j+1], h1 = hh[2*j+1];
                unsigned u = (unsigned)f2bf(g0 * sigmoidf_(g0) * h0)
                           | ((unsigned)f2bf(g1 * sigmoidf_(g1) * h1) << 16);
                *(unsigned*)&GH[base + 2*j] = u;
            }
        }
        __syncthreads();
#pragma unroll
        for (int tl = 0; tl < 2; ++tl) {
            v8f acc = zero8();
#pragma unroll
            for (int ks = 0; ks < 4; ++ks)
                acc = wmma_bf(f1[ks], frag_ld16(GH, ks * 2 + tl), acc);
            frag_stc(OF, w * 2 + tl, acc);
        }
        __syncthreads();
        if (tid < 32) {
            float ss3 = 0.f;
            for (int d = 0; d < 128; ++d) { float v = OF[c_idx(2, d, tid)]; ss3 += v * v; }
            colsc[tid] = rsqrtf(ss3 * (1.f / 128.f) + 1e-5f);
        }
        __syncthreads();
        for (int i = tid; i < 4096; i += 256) {
            int d = i >> 5, l = i & 31;
            int row = b_ * L_ + c * 32 + l;
            onb[(size_t)row * DIM_ + hh2 * HD_ + d] =
                f2bf(OF[c_idx(2, d, l)] * colsc[l] * onw[d]);
        }
        __syncthreads();
    }
}

// ---------------------------------------------------------------------------
// Kernel 4: out = o_n @ w_o^T  (A from onb, B async-staged from row-major w_o)
// ---------------------------------------------------------------------------
__global__ void __launch_bounds__(256) k_out(const unsigned short* __restrict__ on,
                                             const unsigned short* __restrict__ wb,
                                             float* __restrict__ out) {
    extern __shared__ char sm[];
    unsigned short* Als = (unsigned short*)sm;
    unsigned short* Bls = (unsigned short*)(sm + 8192);
    int m0 = blockIdx.x * 128, n0 = blockIdx.y * 128;
    int tid = threadIdx.x, w = tid >> 5;
    v8f acc[8];
#pragma unroll
    for (int t = 0; t < 8; ++t) acc[t] = zero8();

    for (int kt = 0; kt < DIM_; kt += 32) {
#pragma unroll
        for (int p = 0; p < 2; ++p) {
            int i = tid + p * 256;
            int m = i >> 2, oct = i & 3, k8 = oct * 8;
            unsigned adst = 2u * ((unsigned)(((m >> 4) * 32 + (m & 15) + ((oct >> 1) << 4)) * 16
                                             + ((oct & 1) << 3)));
            async_b128(adst, &on[(size_t)(m0 + m) * DIM_ + kt + k8]);
            unsigned bdst = 8192u + 2u * ((unsigned)(((m >> 4) * 32 + (m & 15) + ((oct >> 1) << 4)) * 16
                                                     + ((oct & 1) << 3)));
            async_b128(bdst, &wb[(size_t)(n0 + m) * DIM_ + kt + k8]);
        }
        if (kt + 32 < DIM_) {
            __builtin_prefetch(&on[(size_t)(m0 + (tid >> 1)) * DIM_ + kt + 32]);
            __builtin_prefetch(&wb[(size_t)(n0 + (tid >> 1)) * DIM_ + kt + 32]);
        }
        wait_async0();
        __syncthreads();
        v16bf a = frag_ld16(Als, w);
#pragma unroll
        for (int tc = 0; tc < 8; ++tc)
            acc[tc] = wmma_bf(a, frag_ld16(Bls, tc), acc[tc]);
        __syncthreads();
    }
    int lI = tid & 31, n_ = lI & 15, r8 = (lI >> 4) << 3;
#pragma unroll
    for (int tc = 0; tc < 8; ++tc)
#pragma unroll
        for (int j = 0; j < 8; ++j)
            out[(size_t)(m0 + w * 16 + r8 + j) * DIM_ + n0 + tc * 16 + n_] = acc[tc][j];
}

// ---------------------------------------------------------------------------
// Host-side orchestration
// ---------------------------------------------------------------------------
extern "C" void kernel_launch(void* const* d_in, const int* in_sizes, int n_in,
                              void* d_out, int out_size, void* d_ws, size_t ws_size,
                              hipStream_t stream) {
    (void)in_sizes; (void)n_in; (void)out_size; (void)ws_size;
    const float* x     = (const float*)d_in[0];
    const float* w_qkv = (const float*)d_in[1];
    const float* w_lr  = (const float*)d_in[2];
    const float* w_o   = (const float*)d_in[3];
    const float* w0    = (const float*)d_in[4];
    const float* w1    = (const float*)d_in[5];
    const float* w2    = (const float*)d_in[6];
    const float* onw   = (const float*)d_in[7];
    float* out = (float*)d_out;

    unsigned short* xb    = (unsigned short*)d_ws;               // 8192*2048
    unsigned short* wqkvb = xb    + (size_t)ROWS_ * DIM_;        // 6144*2048 (row-major)
    unsigned short* wob   = wqkvb + (size_t)QKVC_ * DIM_;        // 2048*2048 (row-major)
    unsigned short* qkvh  = wob   + (size_t)DIM_ * DIM_;         // 3*32*4096*128
    unsigned short* onb   = qkvh  + (size_t)3 * BH_ * L_ * HD_;  // 8192*2048
    float*          lrbuf = (float*)(onb + (size_t)ROWS_ * DIM_);// 3*32*4096

    int nx = ROWS_ * DIM_;
    k_cast<<<(nx + 255) / 256, 256, 0, stream>>>(x, xb, nx);
    k_cast<<<(QKVC_ * DIM_ + 255) / 256, 256, 0, stream>>>(w_qkv, wqkvb, QKVC_ * DIM_);
    k_cast<<<(DIM_ * DIM_ + 255) / 256, 256, 0, stream>>>(w_o, wob, DIM_ * DIM_);
    k_lr  <<<(ROWS_ * 48) / 8 / 32, 256, 0, stream>>>(x, w_lr, lrbuf);
    k_qkv <<<dim3(ROWS_ / 128, 48), 256, 16384, stream>>>(xb, wqkvb, qkvh);
    k_head<<<BH_, 256, SMEM_TOTAL, stream>>>(qkvh, lrbuf, w0, w1, w2, onw, onb);
    k_out <<<dim3(ROWS_ / 128, DIM_ / 128), 256, 16384, stream>>>(onb, wob, out);
}
